// ResidualBlock_40226663694521
// MI455X (gfx1250) — matrix operand order repaired and checked by reading, not compiled, not run on hardware
//
#include <hip/hip_runtime.h>
#include <hip/hip_bf16.h>

// Problem constants: x is [T=6, B=40, C=64, H=48, W=48] fp32.
#define NIMG 240            // T*B
#define HH   48
#define WW   48
#define CCH  64
#define PLANE (HH*WW)       // 2304
#define NPIX (NIMG*PLANE)   // 552960 pixels
#define KTOT 576            // 64 ic * 9 taps
#define NBLK_CONV (NIMG*(HH/2))       // 5760 tiles: 64 oc x (2 rows x 48 px)
#define BN_EPS 1e-5f
#define THRESH 0.5f

typedef __attribute__((ext_vector_type(16))) __bf16 v16bf;
typedef __attribute__((ext_vector_type(8)))  float  v8f;
typedef __attribute__((ext_vector_type(4)))  __bf16 bf4;

union Frag { v16bf v; uint4 u[2]; };

// ---------------------------------------------------------------------------
// Kernel 1: NCHW fp32 -> NHWC bf16 via LDS transpose (coalesced both sides)
// ---------------------------------------------------------------------------
__global__ __launch_bounds__(256)
void convert_x_kernel(const float* __restrict__ x, __bf16* __restrict__ xh)
{
    __shared__ __align__(16) __bf16 tile[64 * 72];  // [pixel][72] (pad: 144B rows)
    const int gp0 = blockIdx.x * 64;                // 64 pixels per block
    const int n   = gp0 / PLANE;
    const int p0  = gp0 % PLANE;                    // PLANE % 64 == 0 -> same image

    #pragma unroll
    for (int it = 0; it < 16; ++it) {               // 64ch x 64pix = 4096 loads
        int idx = it * 256 + threadIdx.x;
        int c = idx >> 6, p = idx & 63;
        float v = x[(size_t)(n * CCH + c) * PLANE + p0 + p];
        tile[p * 72 + c] = (__bf16)v;
    }
    __syncthreads();
    #pragma unroll
    for (int it = 0; it < 2; ++it) {                // 64pix x 128B out, 16B chunks
        int idx = it * 256 + threadIdx.x;
        int p = idx >> 3, ck = idx & 7;
        *(uint4*)(xh + (size_t)(gp0 + p) * CCH + ck * 8) =
            *(const uint4*)(tile + p * 72 + ck * 8);
    }
}

// ---------------------------------------------------------------------------
// Kernel 2: both conv weights fp32 [oc][ic][3][3] -> bf16 [oc][tap*64+ic]
// ---------------------------------------------------------------------------
__global__ __launch_bounds__(256)
void convert_w_kernel(const float* __restrict__ w1, const float* __restrict__ w2,
                      __bf16* __restrict__ wh1, __bf16* __restrict__ wh2)
{
    int idx = blockIdx.x * 256 + threadIdx.x;
    if (idx >= 2 * CCH * KTOT) return;
    const float* src = (idx < CCH * KTOT) ? w1 : w2;
    __bf16*      dst = (idx < CCH * KTOT) ? wh1 : wh2;
    int i   = idx % (CCH * KTOT);
    int oc  = i / KTOT;
    int k   = i % KTOT;
    int tap = k >> 6;
    int ic  = k & 63;
    dst[i] = (__bf16)src[(size_t)(oc * CCH + ic) * 9 + tap];
}

// ---------------------------------------------------------------------------
// Kernel 3: implicit-GEMM 3x3 conv with v_wmma_f32_16x16x32_bf16.
// Block = 128 threads = 4 waves; tile = 64 oc x (2 rows x 48 px) = 96 pixels.
// Each wave: one 16-oc M stripe, 6 accumulators (2 rows x 3 sixteen-pixel
// subtiles) so every A (weight) fragment is reused 6x -> ~6x less L2 weight
// traffic. LDS stages the 4x50x64ch zero-padded halo once per block.
// Also emits per-block per-channel (sum, sum^2) partials for the BN stats.
// ---------------------------------------------------------------------------
__global__ __launch_bounds__(128)
void conv3x3_wmma_kernel(const __bf16* __restrict__ act,   // NHWC bf16
                         const __bf16* __restrict__ wgt,   // [64][576] bf16
                         float* __restrict__ outc,         // NHWC (0) / NCHW (1)
                         float* __restrict__ partials,     // [NBLK_CONV][128]
                         int outLayout)
{
    // col stride 144B (50 cols), row stride 7200B (4 rows): 28,800 B total
    __shared__ __align__(32) unsigned char lds[4 * 7200];

    const int blk = blockIdx.x;
    const int n   = blk / (HH / 2);
    const int h   = (blk % (HH / 2)) * 2;
    const int tid = threadIdx.x;

    // Fill halo: 4 rows x 50 cols x 8 chunks of 16B (64 ch bf16 per col)
    for (int idx = tid; idx < 4 * 50 * 8; idx += 128) {
        int row = idx / 400;
        int rem = idx % 400;
        int col = rem >> 3;
        int ck  = rem & 7;
        int hh  = h - 1 + row;          // h-1 .. h+2
        int wc  = col - 1;              // -1 .. 48
        uint4 val = {0u, 0u, 0u, 0u};
        if ((unsigned)hh < (unsigned)HH && (unsigned)wc < (unsigned)WW)
            val = *(const uint4*)(act + (size_t)((n * HH + hh) * WW + wc) * CCH + ck * 8);
        *(uint4*)(lds + row * 7200 + col * 144 + ck * 16) = val;
    }
    __syncthreads();

    const int wave = tid >> 5;          // 0..3 -> oc tile
    const int lane = tid & 31;
    const int m    = lane & 15;         // A: row / B: pixel column
    const int hi   = lane >> 4;         // lane half selects K sub-range
    const int oc0  = wave * 16;

    v8f acc[6] = {};                    // [hr*3 + sub]
    #pragma unroll
    for (int tap = 0; tap < 9; ++tap) {
        const int kh = tap / 3, kw = tap % 3;
        #pragma unroll
        for (int kc = 0; kc < 2; ++kc) {
            // A fragment (16x32) per ISA layout: lanes0-15 K{0..7,16..23},
            // lanes16-31 K{8..15,24..31} relative to k0 = tap*64 + kc*32.
            Frag a;
            const __bf16* ap = wgt + (size_t)(oc0 + m) * KTOT
                             + tap * 64 + kc * 32 + hi * 8;
            a.u[0] = *(const uint4*)(ap);
            a.u[1] = *(const uint4*)(ap + 16);
            #pragma unroll
            for (int hr = 0; hr < 2; ++hr) {
                #pragma unroll
                for (int sub = 0; sub < 3; ++sub) {
                    // B fragment (32x16): lane half hi covers K{hi*8..+7, 16+hi*8..+7}
                    Frag b;
                    const unsigned char* bp = lds + (hr + kh) * 7200
                        + (sub * 16 + m + kw) * 144 + (kc * 32 + hi * 8) * 2;
                    b.u[0] = *(const uint4*)(bp);
                    b.u[1] = *(const uint4*)(bp + 32);
                    acc[hr * 3 + sub] = __builtin_amdgcn_wmma_f32_16x16x32_bf16(
                        false, a.v, false, b.v, (short)0,
                        acc[hr * 3 + sub], false, false);
                }
            }
        }
    }

    // Store conv output (layout per consumer)
    #pragma unroll
    for (int t = 0; t < 6; ++t) {
        const int hr = t / 3, sub = t % 3;
        #pragma unroll
        for (int r = 0; r < 8; ++r) {
            int oc = oc0 + r + 8 * hi;   // D layout: VGPR r, lane half -> M
            size_t gidx;
            if (outLayout == 0)
                gidx = (size_t)((n * HH + h + hr) * WW + (sub * 16 + m)) * CCH + oc;
            else
                gidx = (size_t)(n * CCH + oc) * PLANE + (h + hr) * WW + (sub * 16 + m);
            outc[gidx] = acc[t][r];
        }
    }

    // Per-channel sum / sum^2 over this tile's 96 pixels (16-lane xor reduce)
    float s[8], q[8];
    #pragma unroll
    for (int r = 0; r < 8; ++r) { s[r] = 0.f; q[r] = 0.f; }
    #pragma unroll
    for (int t = 0; t < 6; ++t) {
        #pragma unroll
        for (int r = 0; r < 8; ++r) {
            float v = acc[t][r];
            s[r] += v;
            q[r] += v * v;
        }
    }
    #pragma unroll
    for (int r = 0; r < 8; ++r) {
        #pragma unroll
        for (int msk = 8; msk >= 1; msk >>= 1) {
            s[r] += __shfl_xor(s[r], msk, 16);
            q[r] += __shfl_xor(q[r], msk, 16);
        }
    }
    if (m == 0) {
        float* pb = partials + (size_t)blk * 128;
        #pragma unroll
        for (int r = 0; r < 8; ++r) {
            int oc = oc0 + r + 8 * hi;
            pb[oc]      = s[r];
            pb[64 + oc] = q[r];
        }
    }
}

// ---------------------------------------------------------------------------
// Kernel 4: reduce [NBLK][128] partials -> stats[128] = {sum[64], sumsq[64]}
// ---------------------------------------------------------------------------
__global__ __launch_bounds__(256)
void reduce_stats_kernel(const float* __restrict__ partials,
                         float* __restrict__ stats, int nblk)
{
    __shared__ float sm[256];
    const int col = blockIdx.x;              // 0..127
    float acc = 0.f;
    for (int i = threadIdx.x; i < nblk; i += 256)
        acc += partials[(size_t)i * 128 + col];
    sm[threadIdx.x] = acc;
    __syncthreads();
    for (int s = 128; s > 0; s >>= 1) {
        if (threadIdx.x < s) sm[threadIdx.x] += sm[threadIdx.x + s];
        __syncthreads();
    }
    if (threadIdx.x == 0) stats[col] = sm[0];
}

// ---------------------------------------------------------------------------
// Kernel 5: BN (RMS denom, eps outside sqrt) + spike -> bf16 {0,1} NHWC
// ---------------------------------------------------------------------------
__global__ __launch_bounds__(256)
void bn_spike_kernel(const float* __restrict__ cin,    // NHWC fp32
                     const float* __restrict__ stats,  // [128]
                     const float* __restrict__ gamma,
                     const float* __restrict__ beta,
                     __bf16* __restrict__ sout)        // NHWC bf16
{
    const size_t base = ((size_t)blockIdx.x * 256 + threadIdx.x) * 4;
    const int   c0    = (int)(base & 63);
    const float inv_n = 1.0f / (float)NPIX;
    float4 v = *(const float4*)(cin + base);
    float r[4] = {v.x, v.y, v.z, v.w};
    bf4 o;
    #pragma unroll
    for (int j = 0; j < 4; ++j) {
        int c = c0 + j;
        float mean = stats[c] * inv_n;
        float rms  = sqrtf(stats[64 + c] * inv_n);
        float val  = (r[j] - mean) / (rms + BN_EPS) * gamma[c] + beta[c];
        o[j] = (__bf16)((val > THRESH) ? 1.0f : 0.0f);
    }
    *(bf4*)(sout + base) = o;
}

// ---------------------------------------------------------------------------
// Kernel 6: out = heaviside(bn2(c2) + x - thresh), everything NCHW float4
// ---------------------------------------------------------------------------
__global__ __launch_bounds__(256)
void bn_residual_spike_kernel(const float* __restrict__ c2,   // NCHW fp32
                              const float* __restrict__ x,    // NCHW fp32
                              const float* __restrict__ stats,
                              const float* __restrict__ gamma,
                              const float* __restrict__ beta,
                              float* __restrict__ out)
{
    const size_t base = ((size_t)blockIdx.x * 256 + threadIdx.x) * 4;
    const int c = (int)((base / PLANE) & 63);   // 4 elems stay in one plane
    float mean = stats[c] * (1.0f / (float)NPIX);
    float rms  = sqrtf(stats[64 + c] * (1.0f / (float)NPIX));
    float sc   = gamma[c] / (rms + BN_EPS);
    float sh   = beta[c] - mean * sc;
    float4 v  = *(const float4*)(c2 + base);
    float4 xs = *(const float4*)(x + base);
    float4 o;
    o.x = ((v.x * sc + sh + xs.x) > THRESH) ? 1.0f : 0.0f;
    o.y = ((v.y * sc + sh + xs.y) > THRESH) ? 1.0f : 0.0f;
    o.z = ((v.z * sc + sh + xs.z) > THRESH) ? 1.0f : 0.0f;
    o.w = ((v.w * sc + sh + xs.w) > THRESH) ? 1.0f : 0.0f;
    *(float4*)(out + base) = o;
}

// ---------------------------------------------------------------------------
// Host orchestration
// ---------------------------------------------------------------------------
extern "C" void kernel_launch(void* const* d_in, const int* in_sizes, int n_in,
                              void* d_out, int out_size, void* d_ws, size_t ws_size,
                              hipStream_t stream)
{
    (void)in_sizes; (void)n_in; (void)out_size; (void)ws_size;
    const float* x  = (const float*)d_in[0];
    const float* w1 = (const float*)d_in[1];
    const float* g1 = (const float*)d_in[2];
    const float* b1 = (const float*)d_in[3];
    const float* w2 = (const float*)d_in[4];
    const float* g2 = (const float*)d_in[5];
    const float* b2 = (const float*)d_in[6];
    float* out = (float*)d_out;

    char* ws = (char*)d_ws;
    const size_t SZ_ACT_BF16 = (size_t)NPIX * CCH * 2;   // 70,778,880
    const size_t SZ_ACT_F32  = (size_t)NPIX * CCH * 4;   // 141,557,760
    const size_t SZ_W_BF16   = (size_t)CCH * KTOT * 2;   // 73,728
    __bf16* xh   = (__bf16*)(ws);
    __bf16* s1h  = (__bf16*)(ws + SZ_ACT_BF16);
    float*  cbuf = (float*) (ws + 2 * SZ_ACT_BF16);
    __bf16* wh1  = (__bf16*)(ws + 2 * SZ_ACT_BF16 + SZ_ACT_F32);
    __bf16* wh2  = (__bf16*)(ws + 2 * SZ_ACT_BF16 + SZ_ACT_F32 + SZ_W_BF16);
    float*  partials = (float*)(ws + 2 * SZ_ACT_BF16 + SZ_ACT_F32 + 2 * SZ_W_BF16);
    float*  stats1 = (float*)((char*)partials + (size_t)NBLK_CONV * 128 * 4);
    float*  stats2 = stats1 + 128;

    // Stage 0: layout/precision conversion
    convert_x_kernel<<<NPIX / 64, 256, 0, stream>>>(x, xh);
    convert_w_kernel<<<(2 * CCH * KTOT + 255) / 256, 256, 0, stream>>>(w1, w2, wh1, wh2);

    // Stage 1: conv1 (WMMA) -> c1 NHWC + channel partials -> stats1
    conv3x3_wmma_kernel<<<NBLK_CONV, 128, 0, stream>>>(xh, wh1, cbuf, partials, 0);
    reduce_stats_kernel<<<128, 256, 0, stream>>>(partials, stats1, NBLK_CONV);

    // Stage 2: bn1 + spike -> bf16 spikes NHWC
    bn_spike_kernel<<<(NPIX * CCH / 4) / 256, 256, 0, stream>>>(cbuf, stats1, g1, b1, s1h);

    // Stage 3: conv2 (WMMA) -> c2 NCHW + partials -> stats2
    conv3x3_wmma_kernel<<<NBLK_CONV, 128, 0, stream>>>(s1h, wh2, cbuf, partials, 1);
    reduce_stats_kernel<<<128, 256, 0, stream>>>(partials, stats2, NBLK_CONV);

    // Stage 4: bn2 + residual + spike -> fp32 output NCHW
    bn_residual_spike_kernel<<<(NPIX * CCH / 4) / 256, 256, 0, stream>>>(
        cbuf, x, stats2, g2, b2, out);
}
